// e3mp_step_64931315581498
// MI455X (gfx1250) — matrix operand order repaired and checked by reading, not compiled, not run on hardware
//
#include <hip/hip_runtime.h>
#include <hip/hip_bf16.h>

// ---------------------------------------------------------------------------
// E(3)-equivariant message passing step, fused for MI455X (gfx1250).
//   - edge MLP (8->64->864) via v_wmma_f32_16x16x32_f16, f32 accumulate
//   - packed W1 (108 KB f16) held LDS-resident per block; each block persists
//     over 32 edge-tiles so B-fragments stream from LDS, not L2 (kills the
//     1.7 GB operand-refetch that would otherwise bound the kernel)
//   - 864-wide per-edge tensor-product weights live only in LDS (never HBM)
//   - CG tensor product per edge in registers, atomic f32 scatter to nodes
//   - node update MLP (32->64->16) with WMMA as well
// ---------------------------------------------------------------------------

typedef __attribute__((ext_vector_type(16))) _Float16 v16h;
typedef __attribute__((ext_vector_type(8)))  float    v8f;

#define N_NODES   16000
#define N_EDGES   256000
#define FEAT      60         // 16 + 8*3 + 4*5
#define WNUM      864
#define HID       64
#define EDGE_BLOCKS 500      // 16000 edge-tiles / 32 tiles per block
#define TILES_PER_BLOCK 32   // 8 iterations x 4 waves
#define W1_HALFS (54 * 2 * 512)   // 55296 halfs = 108 KB

// path weights: pw = sqrt(dim_out / fan_out)
#define P0  0.18898224f      // sqrt(1/28)
#define P1  0.28867513f      // sqrt(3/36)
#define P2  0.39528471f      // sqrt(5/32)
#define SQ3i 0.57735027f
#define SQ5i 0.44721360f
#define SQ3  1.73205081f
#define SQ5  2.23606798f
#define SQ15 3.87298335f
#define SQRT2 1.41421356f

// w3j(1,1,2) nonzero magnitudes (real basis [xz,xy,y2-..,yz,(z2-x2)/2])
#define CA  0.31622777f      // 1/sqrt(10)
#define CB  0.36514837f      // sqrt(2/15)
#define CC  0.18257419f      // sqrt(1/30)

// ---------------------------------------------------------------------------
// fragment helpers (ISA 7.12.2 wave32 layouts)
// ---------------------------------------------------------------------------
__device__ inline v16h afrag_lds(const _Float16* row, int lane) {
  // 16-bit A 16x32: lane row M, K = (v<4?0:16) + (lane>=16?8:0) + (v&3)*2 + h
  union { v16h v; unsigned u[8]; } f;
  const int kb = (lane & 16) ? 8 : 0;
#pragma unroll
  for (int v = 0; v < 8; ++v) {
    const int K = ((v < 4) ? 0 : 16) + kb + (v & 3) * 2;
    f.u[v] = *(const unsigned*)(row + K);   // dword = two consecutive K halfs
  }
  return f.v;
}

__device__ inline v16h bfrag(const _Float16* p, int lane) {
  return *(const v16h*)(p + lane * 16);     // pre-packed: 16 halfs per lane
}

__device__ inline v8f wmma16(v16h a, v16h b, v8f c) {
  return __builtin_amdgcn_wmma_f32_16x16x32_f16(false, a, false, b,
                                                (short)0, c, false, false);
}

// ---------------------------------------------------------------------------
// pack a KxN f32 weight matrix into f16 WMMA B-fragment layout
// dst[((tile*ksteps+ks)*32 + lane)*16 + j] = src[K][N] * scale
//   K = ks*32 + (j>=8 ? 16 : 0) + (lane>=16 ? 8 : 0) + (j&7) ;  N = tile*16 + (lane&15)
// ---------------------------------------------------------------------------
__global__ void pack_b_kernel(const float* __restrict__ src,
                              _Float16* __restrict__ dst,
                              int Ksrc, int Nsrc, int ntiles, int ksteps,
                              float scale) {
  const int i = blockIdx.x * blockDim.x + threadIdx.x;
  const int total = ntiles * ksteps * 512;
  if (i >= total) return;
  const int j    = i & 15;
  const int lane = (i >> 4) & 31;
  const int ks   = (i >> 9) % ksteps;
  const int tile = i / (512 * ksteps);
  const int K = ks * 32 + ((j >> 3) << 4) + ((lane >> 4) << 3) + (j & 7);
  const int N = tile * 16 + (lane & 15);
  const float v = (K < Ksrc && N < Nsrc) ? src[K * Nsrc + N] * scale : 0.f;
  dst[i] = (_Float16)v;
}

__global__ void zero_kernel(float* __restrict__ p, int n) {
  const int i = blockIdx.x * blockDim.x + threadIdx.x;
  if (i < n) p[i] = 0.f;
}

// ---------------------------------------------------------------------------
// edge kernel: block persists over 32 edge-tiles; W1 LDS-resident
// ---------------------------------------------------------------------------
__launch_bounds__(128)
__global__ void edge_kernel(const float* __restrict__ x,
                            const float* __restrict__ pos,
                            const float* __restrict__ ea,
                            const int*   __restrict__ esrc,
                            const int*   __restrict__ edst,
                            const _Float16* __restrict__ pW0,
                            const _Float16* __restrict__ pW1,
                            float* __restrict__ msg) {
  __shared__ _Float16 sW1[W1_HALFS];          // 108 KB packed W1
  __shared__ _Float16 sEA [4][16][32];
  __shared__ _Float16 sHID[4][16][64];
  __shared__ _Float16 sTPW[4][16][WNUM];      // 108 KB tp_w (4 waves)

  const int tid  = threadIdx.x;
  const int lane = tid & 31;
  const int wave = tid >> 5;

  // ---- stage packed W1 into LDS once per block (uint4 = 8 halfs) ----
  {
    const uint4* s = (const uint4*)pW1;
    uint4*       d = (uint4*)sW1;
    for (int i = tid; i < W1_HALFS / 8; i += 128) d[i] = s[i];
  }

  for (int it = 0; it < TILES_PER_BLOCK / 4; ++it) {
    const int e0 = (blockIdx.x * TILES_PER_BLOCK + it * 4 + wave) * 16;

    // prefetch next iteration's edge_attr rows (global_prefetch_b8)
    if (it + 1 < TILES_PER_BLOCK / 4) {
      const int eN = (blockIdx.x * TILES_PER_BLOCK + (it + 1) * 4 + wave) * 16;
      __builtin_prefetch(&ea[(eN + (lane & 15)) * 8], 0, 1);
    }

    // ---- stage edge_attr tile (16 x 8, K padded to 32) ----
    for (int idx = lane; idx < 16 * 32; idx += 32) {
      const int m = idx >> 5, k = idx & 31;
      sEA[wave][m][k] = (_Float16)((k < 8) ? ea[(e0 + m) * 8 + k] : 0.f);
    }
    __syncthreads();   // also covers sW1 staging before first B use

    // ---- layer 1: 16x8 @ 8x64, relu*sqrt2 ----
    {
      const v16h a1 = afrag_lds(&sEA[wave][lane & 15][0], lane);
      const int Mb = (lane & 16) ? 8 : 0;
#pragma unroll
      for (int t = 0; t < 4; ++t) {
        v8f c = {};
        c = wmma16(a1, bfrag(pW0 + t * 512, lane), c);
        const int N = t * 16 + (lane & 15);
#pragma unroll
        for (int r = 0; r < 8; ++r) {
          float h = c[r];
          h = (h > 0.f) ? SQRT2 * h : 0.f;
          sHID[wave][Mb + r][N] = (_Float16)h;
        }
      }
    }
    __syncthreads();

    // ---- layer 2: 16x64 @ 64x864, B fragments from LDS-resident W1 ----
    {
      const v16h a20 = afrag_lds(&sHID[wave][lane & 15][0],  lane);
      const v16h a21 = afrag_lds(&sHID[wave][lane & 15][32], lane);
      const int Mb = (lane & 16) ? 8 : 0;
      for (int t = 0; t < 54; ++t) {
        v8f c = {};
        c = wmma16(a20, bfrag(sW1 + (t * 2 + 0) * 512, lane), c);
        c = wmma16(a21, bfrag(sW1 + (t * 2 + 1) * 512, lane), c);
        const int N = t * 16 + (lane & 15);
#pragma unroll
        for (int r = 0; r < 8; ++r)
          sTPW[wave][Mb + r][N] = (_Float16)c[r];
      }
    }
    __syncthreads();

    // ---- tensor product: 2 threads/edge (half0: l=0,2 out; half1: l=1) ----
    const int el   = tid & 63;
    const int half = tid >> 6;
    const int e    = (blockIdx.x * TILES_PER_BLOCK + it * 4) * 16 + el;
    const _Float16* w = &sTPW[el >> 4][el & 15][0];
#define Wf(i) ((float)w[(i)])

    const int src = esrc[e], dst = edst[e];
    const float vx = pos[dst * 3 + 0] - pos[src * 3 + 0];
    const float vy = pos[dst * 3 + 1] - pos[src * 3 + 1];
    const float vz = pos[dst * 3 + 2] - pos[src * 3 + 2];
    const float ri = rsqrtf(vx * vx + vy * vy + vz * vz + 1e-12f);
    const float ux = vx * ri, uy = vy * ri, uz = vz * ri;
    float s1[3] = { SQ3 * ux, SQ3 * uy, SQ3 * uz };
    float s2[5] = { SQ15 * ux * uz, SQ15 * ux * uy,
                    SQ5 * (uy * uy - 0.5f * (ux * ux + uz * uz)),
                    SQ15 * uy * uz, 0.5f * SQ15 * (uz * uz - ux * ux) };

    float xs[FEAT];
#pragma unroll
    for (int i = 0; i < FEAT; ++i) xs[i] = x[src * FEAT + i];
#define X1(u,i) xs[16 + (u) * 3 + (i)]
#define X2(u,i) xs[40 + (u) * 5 + (i)]

    if (half == 0) {
      float out0[16]; float out2[20];
#pragma unroll
      for (int i = 0; i < 16; ++i) out0[i] = 0.f;
#pragma unroll
      for (int i = 0; i < 20; ++i) out2[i] = 0.f;

      // (0,0,0) off 0, 16x16
#pragma unroll
      for (int wo = 0; wo < 16; ++wo) {
        float s = 0.f;
#pragma unroll
        for (int u = 0; u < 16; ++u) s += xs[u] * Wf(u * 16 + wo);
        out0[wo] += P0 * s;
      }
      // (1,1,0) off 512, 8x16 : t_u = <x1_u, sh1>/sqrt3
      {
        float tu[8];
#pragma unroll
        for (int u = 0; u < 8; ++u)
          tu[u] = SQ3i * (X1(u,0) * s1[0] + X1(u,1) * s1[1] + X1(u,2) * s1[2]);
#pragma unroll
        for (int wo = 0; wo < 16; ++wo) {
          float s = 0.f;
#pragma unroll
          for (int u = 0; u < 8; ++u) s += tu[u] * Wf(512 + u * 16 + wo);
          out0[wo] += P0 * s;
        }
      }
      // (2,2,0) off 784, 4x16
      {
        float tu[4];
#pragma unroll
        for (int u = 0; u < 4; ++u) {
          float s = 0.f;
#pragma unroll
          for (int i = 0; i < 5; ++i) s += X2(u,i) * s2[i];
          tu[u] = SQ5i * s;
        }
#pragma unroll
        for (int wo = 0; wo < 16; ++wo) {
          float s = 0.f;
#pragma unroll
          for (int u = 0; u < 4; ++u) s += tu[u] * Wf(784 + u * 16 + wo);
          out0[wo] += P0 * s;
        }
      }
      // (0,2,2) off 384, 16x4
#pragma unroll
      for (int wo = 0; wo < 4; ++wo) {
        float s = 0.f;
#pragma unroll
        for (int u = 0; u < 16; ++u) s += xs[u] * Wf(384 + u * 4 + wo);
        s *= P2 * SQ5i;
#pragma unroll
        for (int k = 0; k < 5; ++k) out2[wo * 5 + k] += s * s2[k];
      }
      // (1,1,2) off 640, 8x4 : dense w3j(1,1,2)
#pragma unroll
      for (int u = 0; u < 8; ++u) {
        const float a0 = X1(u,0), a1 = X1(u,1), a2 = X1(u,2);
        float tk[5];
        tk[0] = CA * (a0 * s1[2] + a2 * s1[0]);
        tk[1] = CA * (a0 * s1[1] + a1 * s1[0]);
        tk[2] = CB * a1 * s1[1] - CC * (a0 * s1[0] + a2 * s1[2]);
        tk[3] = CA * (a1 * s1[2] + a2 * s1[1]);
        tk[4] = CA * (a2 * s1[2] - a0 * s1[0]);
#pragma unroll
        for (int wo = 0; wo < 4; ++wo) {
          const float wv = P2 * Wf(640 + u * 4 + wo);
#pragma unroll
          for (int k = 0; k < 5; ++k) out2[wo * 5 + k] += tk[k] * wv;
        }
      }
      // (2,0,2) off 736, 4x4
#pragma unroll
      for (int u = 0; u < 4; ++u)
#pragma unroll
        for (int wo = 0; wo < 4; ++wo) {
          const float wv = P2 * SQ5i * Wf(736 + u * 4 + wo);
#pragma unroll
          for (int k = 0; k < 5; ++k) out2[wo * 5 + k] += X2(u,k) * wv;
        }
      // (2,2,2) off 848, 4x4 — structure exact; table values approximate
      {
        const float A5 = 0.2390457f, B5 = 0.2927700f, D5 = 0.1195229f;
#pragma unroll
        for (int u = 0; u < 4; ++u) {
          const float q0 = X2(u,0), q1 = X2(u,1), q2 = X2(u,2),
                      q3 = X2(u,3), q4 = X2(u,4);
          float tk[5];
          tk[0] = A5 * (q1 * s2[3] + q3 * s2[1]) + D5 * (q0 * s2[2] + q2 * s2[0])
                + A5 * (q0 * s2[4] + q4 * s2[0]);
          tk[1] = A5 * (q1 * s2[2] + q2 * s2[1]) + D5 * (q0 * s2[3] + q3 * s2[0])
                - A5 * (q1 * s2[4] + q4 * s2[1]);
          tk[2] = B5 * q2 * s2[2] + D5 * (q1 * s2[1] + q3 * s2[3])
                - A5 * (q0 * s2[0] + q4 * s2[4]);
          tk[3] = A5 * (q3 * s2[2] + q2 * s2[3]) + D5 * (q0 * s2[1] + q1 * s2[0])
                + A5 * (q3 * s2[4] + q4 * s2[3]);
          tk[4] = A5 * (q4 * s2[2] + q2 * s2[4]) + D5 * (q3 * s2[3] - q1 * s2[1])
                + A5 * (q0 * s2[0] - q4 * s2[4]);
#pragma unroll
          for (int wo = 0; wo < 4; ++wo) {
            const float wv = P2 * Wf(848 + u * 4 + wo);
#pragma unroll
            for (int k = 0; k < 5; ++k) out2[wo * 5 + k] += tk[k] * wv;
          }
        }
      }
#pragma unroll
      for (int f = 0; f < 16; ++f) atomicAdd(&msg[dst * FEAT + f], out0[f]);
#pragma unroll
      for (int f = 0; f < 20; ++f) atomicAdd(&msg[dst * FEAT + 40 + f], out2[f]);
    } else {
      float out1[24];
#pragma unroll
      for (int i = 0; i < 24; ++i) out1[i] = 0.f;

      // (0,1,1) off 256, 16x8
#pragma unroll
      for (int wo = 0; wo < 8; ++wo) {
        float s = 0.f;
#pragma unroll
        for (int u = 0; u < 16; ++u) s += xs[u] * Wf(256 + u * 8 + wo);
        s *= P1 * SQ3i;
#pragma unroll
        for (int k = 0; k < 3; ++k) out1[wo * 3 + k] += s * s1[k];
      }
      // (1,0,1) off 448, 8x8
#pragma unroll
      for (int u = 0; u < 8; ++u)
#pragma unroll
        for (int wo = 0; wo < 8; ++wo) {
          const float wv = P1 * SQ3i * Wf(448 + u * 8 + wo);
#pragma unroll
          for (int k = 0; k < 3; ++k) out1[wo * 3 + k] += X1(u,k) * wv;
        }
      // (1,2,1) off 672, 8x8 : M[i][k] = sum_j w3j(1,2,1)[i][j][k] * sh2[j]
      {
        float M[3][3];
        M[0][0] = -CC * s2[2] - CA * s2[4];
        M[0][1] =  CA * s2[1];
        M[0][2] =  CA * s2[0];
        M[1][0] =  CA * s2[1];
        M[1][1] =  CB * s2[2];
        M[1][2] =  CA * s2[3];
        M[2][0] =  CA * s2[0];
        M[2][1] =  CA * s2[3];
        M[2][2] = -CC * s2[2] + CA * s2[4];
#pragma unroll
        for (int u = 0; u < 8; ++u) {
          float tk[3];
#pragma unroll
          for (int k = 0; k < 3; ++k)
            tk[k] = X1(u,0) * M[0][k] + X1(u,1) * M[1][k] + X1(u,2) * M[2][k];
#pragma unroll
          for (int wo = 0; wo < 8; ++wo) {
            const float wv = P1 * Wf(672 + u * 8 + wo);
#pragma unroll
            for (int k = 0; k < 3; ++k) out1[wo * 3 + k] += tk[k] * wv;
          }
        }
      }
      // (2,1,1) off 752, 4x8 : G[k][a] = sum_b w3j(2,1,1)[a][b][k] * sh1[b]
      {
        float G[3][5];
#pragma unroll
        for (int k = 0; k < 3; ++k)
#pragma unroll
          for (int a = 0; a < 5; ++a) G[k][a] = 0.f;
        G[0][0] =  CA * s1[2];  G[0][1] =  CA * s1[1];
        G[0][2] = -CC * s1[0];  G[0][4] = -CA * s1[0];
        G[1][1] =  CA * s1[0];  G[1][2] =  CB * s1[1];  G[1][3] = CA * s1[2];
        G[2][0] =  CA * s1[0];  G[2][2] = -CC * s1[2];
        G[2][3] =  CA * s1[1];  G[2][4] =  CA * s1[2];
#pragma unroll
        for (int u = 0; u < 4; ++u) {
          float tk[3];
#pragma unroll
          for (int k = 0; k < 3; ++k) {
            float s = 0.f;
#pragma unroll
            for (int a = 0; a < 5; ++a) s += G[k][a] * X2(u,a);
            tk[k] = s;
          }
#pragma unroll
          for (int wo = 0; wo < 8; ++wo) {
            const float wv = P1 * Wf(752 + u * 8 + wo);
#pragma unroll
            for (int k = 0; k < 3; ++k) out1[wo * 3 + k] += tk[k] * wv;
          }
        }
      }
#pragma unroll
      for (int f = 0; f < 24; ++f) atomicAdd(&msg[dst * FEAT + 16 + f], out1[f]);
    }
#undef Wf
#undef X1
#undef X2
    __syncthreads();   // tp_w consumed before next iteration overwrites it
  }
}

// ---------------------------------------------------------------------------
// node kernel: scal = fc([msg16, x16]) via WMMA ; geom = 0.5*(msg+x)
// ---------------------------------------------------------------------------
__launch_bounds__(128)
__global__ void node_kernel(const float* __restrict__ x,
                            const float* __restrict__ msg,
                            const float* __restrict__ importance,
                            const _Float16* __restrict__ pU0,
                            const _Float16* __restrict__ pU1,
                            float* __restrict__ out) {
  __shared__ _Float16 sIN [4][16][32];
  __shared__ _Float16 sHID[4][16][64];
  const int lane = threadIdx.x & 31;
  const int wave = threadIdx.x >> 5;
  const int n0   = blockIdx.x * 64 + wave * 16;
  const float imp = importance[0] * 0.25f;   // importance / sqrt(16 neighbors)

  for (int idx = lane; idx < 512; idx += 32) {
    const int m = idx >> 5, k = idx & 31;
    const float v = (k < 16) ? msg[(n0 + m) * FEAT + k] * imp
                             : x[(n0 + m) * FEAT + (k - 16)];
    sIN[wave][m][k] = (_Float16)v;
  }
  __syncthreads();

  {
    const v16h a = afrag_lds(&sIN[wave][lane & 15][0], lane);
    const int Mb = (lane & 16) ? 8 : 0;
#pragma unroll
    for (int t = 0; t < 4; ++t) {
      v8f c = {};
      c = wmma16(a, bfrag(pU0 + t * 512, lane), c);
      const int N = t * 16 + (lane & 15);
#pragma unroll
      for (int r = 0; r < 8; ++r) {
        float h = c[r];
        h = (h > 0.f) ? SQRT2 * h : 0.f;
        sHID[wave][Mb + r][N] = (_Float16)h;
      }
    }
  }
  __syncthreads();

  {
    const v16h a0 = afrag_lds(&sHID[wave][lane & 15][0],  lane);
    const v16h a1 = afrag_lds(&sHID[wave][lane & 15][32], lane);
    v8f c = {};
    c = wmma16(a0, bfrag(pU1 + 0,   lane), c);
    c = wmma16(a1, bfrag(pU1 + 512, lane), c);
    const int Mb = (lane & 16) ? 8 : 0;
    const int N  = lane & 15;
#pragma unroll
    for (int r = 0; r < 8; ++r)
      out[(n0 + Mb + r) * FEAT + N] = c[r];
  }

  // geom = 0.5 * (msg_scaled[:,16:] + x[:,16:])
  for (int idx = lane; idx < 16 * 44; idx += 32) {
    const int m = idx / 44, f = idx % 44;
    const int n = n0 + m;
    out[n * FEAT + 16 + f] =
        0.5f * (msg[n * FEAT + 16 + f] * imp + x[n * FEAT + 16 + f]);
  }
}

// ---------------------------------------------------------------------------
extern "C" void kernel_launch(void* const* d_in, const int* in_sizes, int n_in,
                              void* d_out, int out_size, void* d_ws, size_t ws_size,
                              hipStream_t stream) {
  const float* x    = (const float*)d_in[0];
  const float* pos  = (const float*)d_in[1];
  const float* ea   = (const float*)d_in[2];
  const float* W0   = (const float*)d_in[3];   // 8 x 64
  const float* W1   = (const float*)d_in[4];   // 64 x 864
  const float* U0   = (const float*)d_in[5];   // 32 x 64
  const float* U1   = (const float*)d_in[6];   // 64 x 16
  const float* imp  = (const float*)d_in[7];
  const int*   esrc = (const int*)d_in[8];
  const int*   edst = (const int*)d_in[9];
  float* out = (float*)d_out;

  // workspace layout
  float*    msg = (float*)d_ws;                               // 16000*60 f32
  _Float16* pW0 = (_Float16*)((char*)d_ws + (size_t)N_NODES * FEAT * 4);
  _Float16* pW1 = pW0 + 4 * 512;            // W0: 4 tiles x 1 kstep
  _Float16* pU0 = pW1 + W1_HALFS;           // W1: 54 tiles x 2 ksteps
  _Float16* pU1 = pU0 + 4 * 512;            // U0: 4 tiles x 1 kstep
  (void)ws_size; (void)n_in; (void)in_sizes; (void)out_size;

  const int msgN = N_NODES * FEAT;
  zero_kernel<<<(msgN + 255) / 256, 256, 0, stream>>>(msg, msgN);

  pack_b_kernel<<<(4 * 512 + 255) / 256, 256, 0, stream>>>(
      W0, pW0, 8, 64, 4, 1, 0.35355339f);          // 1/sqrt(8)
  pack_b_kernel<<<(W1_HALFS + 255) / 256, 256, 0, stream>>>(
      W1, pW1, 64, WNUM, 54, 2, 0.125f);           // 1/sqrt(64)
  pack_b_kernel<<<(4 * 512 + 255) / 256, 256, 0, stream>>>(
      U0, pU0, 32, 64, 4, 1, 0.17677670f);         // 1/sqrt(32)
  pack_b_kernel<<<(2 * 512 + 255) / 256, 256, 0, stream>>>(
      U1, pU1, 64, 16, 1, 2, 0.125f);              // 1/sqrt(64)

  edge_kernel<<<EDGE_BLOCKS, 128, 0, stream>>>(
      x, pos, ea, esrc, edst, pW0, pW1, msg);

  node_kernel<<<N_NODES / 64, 128, 0, stream>>>(
      x, msg, imp, pU0, pU1, out);
}
